// GCN_28046136442917
// MI455X (gfx1250) — matrix operand order repaired and checked by reading, not compiled, not run on hardware
//
#include <hip/hip_runtime.h>
#include <hip/hip_bf16.h>

// ---------------------------------------------------------------------------
// GCN (2-layer, GCNConv + BatchNorm1d) for MI455X / gfx1250, wave32 + WMMA.
//
// Feature-major formulation:  G^T (F x N) = M^T (F x N) @ Ahat (N x N),
//   Ahat[j,i] = (adj[j,i] + (i==j)) * dinv[j] * dinv[i]
// The 604MB fp32 adjacency streams once per layer as the row-major B operand,
// converted to bf16 in registers (v_cvt_pk_bf16_f32) and fed to
// v_wmma_f32_16x16x32_bf16. Biases b1/b2 cancel under BatchNorm.
// ---------------------------------------------------------------------------

typedef __attribute__((ext_vector_type(16))) __bf16 v16bf;
typedef __attribute__((ext_vector_type(8)))  float  v8f;

union Frag { uint4 u[2]; v16bf v; };

__device__ __forceinline__ unsigned pack2_bf16(float lo, float hi) {
  union { __bf16 h[2]; unsigned u; } p;
  p.h[0] = (__bf16)lo; p.h[1] = (__bf16)hi;   // -> v_cvt_pk_bf16_f32
  return p.u;
}

// -------- init: deg=1 (self loop), zero split-K accumulators ---------------
__global__ void gcn_init(float* __restrict__ deg, float* __restrict__ G1,
                         float* __restrict__ G2, int N) {
  int t = blockIdx.x * blockDim.x + threadIdx.x;
  if (t < N) deg[t] = 1.0f;
  if (t < 128 * N) G1[t] = 0.0f;
  if (t < 64 * N)  G2[t] = 0.0f;
}

// -------- column sums of adj (row-split, atomic accumulate) ----------------
__global__ void gcn_colsum(const float* __restrict__ adj, float* __restrict__ deg,
                           int N, int rows_per) {
  int j  = blockIdx.x * blockDim.x + threadIdx.x;
  int i0 = blockIdx.y * rows_per;
  float s = 0.0f;
  for (int i = i0; i < i0 + rows_per; ++i) s += adj[(size_t)i * N + j];
  atomicAdd(&deg[j], s);
}

__global__ void gcn_dinv(float* __restrict__ d, int N) {
  int j = blockIdx.x * blockDim.x + threadIdx.x;
  if (j < N) { float v = d[j]; d[j] = (v > 0.0f) ? rsqrtf(v) : 0.0f; }
}

// -------- M1 = (x @ W1)^T  (128 x N), bf16 --------------------------------
__global__ void gcn_m1(const float* __restrict__ x, const float* __restrict__ W1,
                       __bf16* __restrict__ M1, int N) {
  int t = blockIdx.x * blockDim.x + threadIdx.x;   // f fastest -> W1 coalesced
  int f = t & 127, i = t >> 7;
  if (i >= N) return;
  const float* xr = x + (size_t)i * 256;
  float acc = 0.0f;
#pragma unroll 8
  for (int c = 0; c < 256; ++c) acc += xr[c] * W1[c * 128 + f];
  M1[(size_t)f * N + i] = (__bf16)acc;
}

// -------- M2 = W2^T @ H1  (64 x N), bf16 ----------------------------------
__global__ void gcn_m2(const float* __restrict__ H1, const float* __restrict__ W2,
                       __bf16* __restrict__ M2, int N) {
  int t = blockIdx.x * blockDim.x + threadIdx.x;
  int f2 = t & 63, i = t >> 6;
  if (i >= N) return;
  float acc = 0.0f;
#pragma unroll 8
  for (int f = 0; f < 128; ++f) acc += H1[(size_t)f * N + i] * W2[f * 64 + f2];
  M2[(size_t)f2 * N + i] = (__bf16)acc;
}

// -------- aggregation GEMM: G (FR x N) += Mb (FR x N) @ Ahat ---------------
// Block tile: FR rows x 128 cols; wave tile 32x64 = 2x4 WMMA 16x16 tiles.
// Split-K over grid.y with fp32 atomics into zeroed G.
template<int FR, int SPLITK>
__global__ __launch_bounds__(FR * 2)
void gcn_agg_wmma(const __bf16* __restrict__ Mb, const float* __restrict__ adj,
                  const float* __restrict__ dinv, float* __restrict__ G, int N) {
  constexpr int NT = 128, KS = 32, NTHR = FR * 2;
  constexpr int ACH = (FR * 4) / NTHR;            // 16B A-chunks per thread (=2)
  constexpr int BCH = (16 * (NT / 4)) / NTHR;     // 2row x 4col B-chunks (2 or 4)

  const int n0     = blockIdx.x * NT;
  const int kchunk = N / SPLITK;
  const int kbeg   = blockIdx.y * kchunk;
  const int kend   = kbeg + kchunk;

  // 80-byte padded rows (40 halves): 16B-aligned runs, conflict-free banks
  __shared__ __align__(16) __bf16 Als[FR * 40];   // A tile: [f][k]
  __shared__ __align__(16) __bf16 Bts[NT * 40];   // B tile transposed: [n][k]
  __shared__ float dn[NT];

  const int tid = threadIdx.x;
  for (int q = tid; q < NT; q += NTHR) dn[q] = dinv[n0 + q];

  const int lane = tid & 31, wave = tid >> 5;
  const int wn = wave & 1, wf = wave >> 1;        // 64-col / 32-row wave tiles
  const int lr = lane & 15, hi = lane >> 4;

  // per-thread constant offsets (uniform base pointers advance per K-step)
  size_t aOff[ACH]; int aLds[ACH];
#pragma unroll
  for (int t = 0; t < ACH; ++t) {
    int c = tid + t * NTHR;                       // 0 .. FR*4-1
    int row = c >> 2, off = (c & 3) * 8;
    aOff[t] = (size_t)row * N + off;
    aLds[t] = row * 40 + off;
  }
  size_t bOff[BCH]; int kkA[BCH], nqA[BCH];
#pragma unroll
  for (int t = 0; t < BCH; ++t) {
    int c = tid + t * NTHR;                       // 0 .. 511
    int kk = (c >> 5) * 2, nq = (c & 31) * 4;     // 2 k-rows x 4 cols per chunk
    bOff[t] = (size_t)kk * N + n0 + nq;
    kkA[t] = kk; nqA[t] = nq;
  }

  v8f acc[2][4] = {};

  const __bf16* mbRow  = Mb + kbeg;
  const float*  adjRow = adj + (size_t)kbeg * N;
  const float*  dvRow  = dinv + kbeg;
  const size_t  adjStep = (size_t)KS * N;

  __syncthreads();

  for (int k0 = kbeg; k0 < kend; k0 += KS) {
    const bool havenext = (k0 + KS) < kend;       // uniform

    // ---- stage A tile (FR x 32 bf16) ----
#pragma unroll
    for (int t = 0; t < ACH; ++t)
      *(uint4*)(Als + aLds[t]) = *(const uint4*)(mbRow + aOff[t]);

    // ---- stage B tile: 2 adj rows -> scaled bf16 pairs, transposed ----
#pragma unroll
    for (int t = 0; t < BCH; ++t) {
      const float* p = adjRow + bOff[t];
      float4 r0 = *(const float4*)p;
      float4 r1 = *(const float4*)(p + N);
      if (havenext) {
        __builtin_prefetch(p + adjStep, 0, 3);
        __builtin_prefetch(p + adjStep + N, 0, 3);
      }
      float2 dk = *(const float2*)(dvRow + kkA[t]);
      const int kr = k0 + kkA[t];                 // global row of r0
      const int nc = n0 + nqA[t];
      float e0[4] = {r0.x, r0.y, r0.z, r0.w};
      float e1[4] = {r1.x, r1.y, r1.z, r1.w};
      __bf16* dst = Bts + kkA[t];
#pragma unroll
      for (int q = 0; q < 4; ++q) {               // branchless self-loop add
        float lo = (e0[q] + ((kr     == nc + q) ? 1.0f : 0.0f)) * dk.x * dn[nqA[t] + q];
        float hh = (e1[q] + ((kr + 1 == nc + q) ? 1.0f : 0.0f)) * dk.y * dn[nqA[t] + q];
        *(unsigned*)(dst + (nqA[t] + q) * 40) = pack2_bf16(lo, hh);
      }
    }
    __syncthreads();

    // ---- fragments per ISA 7.12.2 16-bit layouts (two 16B reads each) ----
    Frag a[2], b[4];
#pragma unroll
    for (int mt = 0; mt < 2; ++mt) {
      const uint4* rp = (const uint4*)(Als + (wf * 32 + mt * 16 + lr) * 40);
      a[mt].u[0] = rp[hi]; a[mt].u[1] = rp[2 + hi];
    }
#pragma unroll
    for (int nt = 0; nt < 4; ++nt) {
      const uint4* cp = (const uint4*)(Bts + (wn * 64 + nt * 16 + lr) * 40);
      b[nt].u[0] = cp[hi]; b[nt].u[1] = cp[2 + hi];
    }
#pragma unroll
    for (int mt = 0; mt < 2; ++mt)
#pragma unroll
      for (int nt = 0; nt < 4; ++nt)
        acc[mt][nt] = __builtin_amdgcn_wmma_f32_16x16x32_bf16(
            false, a[mt].v, false, b[nt].v, (short)0, acc[mt][nt], false, false);
    __syncthreads();

    mbRow  += KS;
    adjRow += adjStep;
    dvRow  += KS;
  }

  // ---- split-K reduce: C/D layout VGPR r -> (M = hi*8+r, N = lr) ----
  const int fB = wf * 32 + hi * 8;
  const int nB = n0 + wn * 64 + lr;
#pragma unroll
  for (int mt = 0; mt < 2; ++mt)
#pragma unroll
    for (int nt = 0; nt < 4; ++nt)
#pragma unroll
      for (int r = 0; r < 8; ++r)
        atomicAdd(&G[(size_t)(fB + mt * 16 + r) * N + nB + nt * 16],
                  acc[mt][nt][r]);
}

// -------- BatchNorm1d (training stats) + optional ReLU / transpose store ---
template<bool RELU, bool TR>
__global__ __launch_bounds__(256)
void gcn_bn(float* __restrict__ G, const float* __restrict__ gamma,
            const float* __restrict__ beta, float* __restrict__ out,
            int F, int N) {
  const int f = blockIdx.x, tid = threadIdx.x;
  float* row = G + (size_t)f * N;
  float s = 0.0f, sq = 0.0f;
  for (int i = tid; i < N; i += 256) { float v = row[i]; s += v; sq += v * v; }
  __shared__ float rs[256], rq[256];
  rs[tid] = s; rq[tid] = sq; __syncthreads();
  for (int o = 128; o > 0; o >>= 1) {
    if (tid < o) { rs[tid] += rs[tid + o]; rq[tid] += rq[tid + o]; }
    __syncthreads();
  }
  const float mu  = rs[0] / (float)N;
  const float var = rq[0] / (float)N - mu * mu;   // biased, like jnp.var
  const float sc  = rsqrtf(var + 1e-5f) * gamma[f];
  const float sh  = beta[f] - mu * sc;
  for (int i = tid; i < N; i += 256) {
    float v = row[i] * sc + sh;
    if (RELU) v = fmaxf(v, 0.0f);
    if (TR) out[(size_t)i * F + f] = v;   // final (N x F) row-major output
    else    row[i] = v;                   // in-place H1
  }
}

// ---------------------------------------------------------------------------
extern "C" void kernel_launch(void* const* d_in, const int* in_sizes, int n_in,
                              void* d_out, int out_size, void* d_ws, size_t ws_size,
                              hipStream_t stream) {
  (void)n_in; (void)out_size; (void)ws_size;
  const float* x      = (const float*)d_in[0];
  const float* adj    = (const float*)d_in[1];
  const float* W1     = (const float*)d_in[2];
  // d_in[3] = b1 : per-feature constant before BatchNorm -> cancels exactly
  const float* gamma1 = (const float*)d_in[4];
  const float* beta1  = (const float*)d_in[5];
  const float* W2     = (const float*)d_in[6];
  // d_in[7] = b2 : cancels likewise
  const float* gamma2 = (const float*)d_in[8];
  const float* beta2  = (const float*)d_in[9];
  const int N = in_sizes[0] / 256;   // 12288

  // workspace carve-out (~14.3 MB)
  size_t off = 0;
  auto alloc = [&](size_t bytes) {
    off = (off + 255) & ~(size_t)255;
    void* p = (char*)d_ws + off;
    off += bytes;
    return p;
  };
  float*  dinv = (float*) alloc((size_t)N * 4);
  __bf16* M1b  = (__bf16*)alloc((size_t)128 * N * 2);
  float*  G1   = (float*) alloc((size_t)128 * N * 4);
  __bf16* M2b  = (__bf16*)alloc((size_t)64  * N * 2);
  float*  G2   = (float*) alloc((size_t)64  * N * 4);

  gcn_init  <<<(128 * N + 255) / 256, 256, 0, stream>>>(dinv, G1, G2, N);
  gcn_colsum<<<dim3(N / 256, 16),     256, 0, stream>>>(adj, dinv, N, N / 16);
  gcn_dinv  <<<(N + 255) / 256,       256, 0, stream>>>(dinv, N);

  // layer 1
  gcn_m1<<<(128 * N) / 256, 256, 0, stream>>>(x, W1, M1b, N);
  gcn_agg_wmma<128, 8><<<dim3(N / 128, 8), 256, 0, stream>>>(M1b, adj, dinv, G1, N);
  gcn_bn<true, false><<<128, 256, 0, stream>>>(G1, gamma1, beta1, nullptr, 128, N);

  // layer 2
  gcn_m2<<<(64 * N) / 256, 256, 0, stream>>>(G1, W2, M2b, N);
  gcn_agg_wmma<64, 8><<<dim3(N / 128, 8), 128, 0, stream>>>(M2b, adj, dinv, G2, N);
  gcn_bn<false, true><<<64, 256, 0, stream>>>(G2, gamma2, beta2, (float*)d_out, 64, N);
}
